// GPT_24567212933134
// MI455X (gfx1250) — matrix operand order repaired and checked by reading, not compiled, not run on hardware
//
#include <hip/hip_runtime.h>
#include <math.h>

// ---- model dims (GPT-2 small, B=2, T=1024) ----
#define V_      50257
#define TBLK    1024
#define NL      8
#define NH      12
#define DMODEL  768
#define HDIM    64
#define NTOK    2048          // B*T
#define EPS_LN  1e-5f

typedef __attribute__((ext_vector_type(16))) __bf16 v16bf;
typedef __attribute__((ext_vector_type(8)))  float  v8f;

struct __align__(16) U4 { unsigned int x, y, z, w; };

// fp32 -> bf16 round-to-nearest-even
static __device__ __forceinline__ unsigned short f2bf(float f) {
  union { float f; unsigned int u; } c; c.f = f;
  unsigned int r = c.u + 0x7fffu + ((c.u >> 16) & 1u);
  return (unsigned short)(r >> 16);
}

static __device__ __forceinline__ float gelu_f(float x) {
  const float k = 0.7978845608028654f;   // sqrt(2/pi)
  float t = tanhf(k * (x + 0.044715f * x * x * x));
  return 0.5f * x * (1.0f + t);
}

// ---------------- embedding: x = wte[idx] + wpe[t] ----------------
__global__ void embed_kernel(const int* __restrict__ idx,
                             const float* __restrict__ wte,
                             const float* __restrict__ wpe,
                             float* __restrict__ x) {
  int tok = blockIdx.x;
  const float* wrow = wte + (size_t)idx[tok] * DMODEL;
  const float* prow = wpe + (size_t)(tok % TBLK) * DMODEL;
  float* xr = x + (size_t)tok * DMODEL;
  for (int c = threadIdx.x; c < DMODEL; c += blockDim.x)
    xr[c] = wrow[c] + prow[c];
}

// ---------------- layernorm: f32 in, bf16 out (one block per row) ----------------
__global__ void layernorm_kernel(const float* __restrict__ x,
                                 const float* __restrict__ s,
                                 const float* __restrict__ b,
                                 unsigned short* __restrict__ out) {
  __shared__ float red[256];
  int tok = blockIdx.x;
  const float* xr = x + (size_t)tok * DMODEL;
  float ls = 0.f;
  for (int c = threadIdx.x; c < DMODEL; c += 256) ls += xr[c];
  red[threadIdx.x] = ls; __syncthreads();
  for (int o = 128; o > 0; o >>= 1) {
    if (threadIdx.x < o) red[threadIdx.x] += red[threadIdx.x + o];
    __syncthreads();
  }
  float mu = red[0] * (1.0f / DMODEL);
  __syncthreads();
  float lv = 0.f;
  for (int c = threadIdx.x; c < DMODEL; c += 256) {
    float d = xr[c] - mu; lv += d * d;
  }
  red[threadIdx.x] = lv; __syncthreads();
  for (int o = 128; o > 0; o >>= 1) {
    if (threadIdx.x < o) red[threadIdx.x] += red[threadIdx.x + o];
    __syncthreads();
  }
  float rstd = rsqrtf(red[0] * (1.0f / DMODEL) + EPS_LN);
  unsigned short* orow = out + (size_t)tok * DMODEL;
  for (int c = threadIdx.x; c < DMODEL; c += 256)
    orow[c] = f2bf((xr[c] - mu) * rstd * s[c] + b[c]);
}

// ------- weight prep: fp32 [K,N] -> bf16 [N,K] (LDS tile transpose) -------
__global__ void convT_bf16(const float* __restrict__ src, unsigned short* __restrict__ dst,
                           int K, int N) {
  __shared__ unsigned short t[32][33];
  int kb = blockIdx.x * 32, nb = blockIdx.y * 32;
  int lr = threadIdx.x >> 5;     // 0..7
  int lc = threadIdx.x & 31;     // 0..31
#pragma unroll
  for (int i = 0; i < 32; i += 8) {
    int n = nb + lc;
    t[lr + i][lc] = (n < N) ? f2bf(src[(size_t)(kb + lr + i) * N + n]) : (unsigned short)0;
  }
  __syncthreads();
#pragma unroll
  for (int i = 0; i < 32; i += 8) {
    int n = nb + lr + i;
    if (n < N) dst[(size_t)n * K + kb + lc] = t[lc][lr + i];
  }
}

// ------- elementwise fp32 -> bf16 (wte is already [N,K]) -------
__global__ void conv_bf16(const float* __restrict__ src, unsigned short* __restrict__ dst,
                          long n) {
  long i = (long)blockIdx.x * blockDim.x + threadIdx.x;
  long st = (long)gridDim.x * blockDim.x;
  for (; i < n; i += st) dst[i] = f2bf(src[i]);
}

// ---------------- WMMA bf16 GEMM ----------------
// C[M,N] = act( A[M,K](bf16) @ W[N,K](bf16)^T + bias ) (+ residual)
// 256 threads = 8 wave32 waves; 128x64 block tile; BK=32; LDS ping-pong.
// Wave owns a 32x32 strip -> 4x v_wmma_f32_16x16x32_bf16 per K-step.
template<bool GELU, bool RES, bool BIAS, bool OUTBF>
__global__ void gemm_wmma(const unsigned short* __restrict__ A,
                          const unsigned short* __restrict__ W,
                          const float* __restrict__ bias,
                          const float* __restrict__ Res,
                          void* __restrict__ Cv,
                          int M, int N, int K) {
  __shared__ unsigned short As[2][128][40];   // [buf][m][k], +16B pad
  __shared__ unsigned short Bs[2][64][40];    // [buf][n][k], +16B pad

  const int tid  = threadIdx.x;
  const int lane = tid & 31;
  const int wid  = tid >> 5;
  const int wr   = wid & 3;        // 32-row band within 128
  const int wc   = wid >> 2;       // 32-col band within 64
  const int m0   = blockIdx.y * 128;
  const int n0   = blockIdx.x * 64;
  const int half = lane >> 4;
  const int l16  = lane & 15;

  v8f acc00 = {}, acc01 = {}, acc10 = {}, acc11 = {};

  // staging mapping: A -> 16 bf16 / thread, B -> 8 bf16 / thread
  const int sar = tid >> 1;                 // 0..127
  const int sac = (tid & 1) * 16;           // 0 or 16
  const int sbr = tid >> 2;                 // 0..63
  const int sbc = (tid & 3) * 8;            // 0,8,16,24
  const unsigned short* Abase = A + (size_t)(m0 + sar) * K + sac;
  const unsigned short* Wbase = W + (size_t)(n0 + sbr) * K + sbc;
  const bool wok = (n0 + sbr) < N;

  auto stage = [&](int p, int kk) {
    const U4* ap = (const U4*)(Abase + kk);
    U4 a0 = ap[0], a1 = ap[1];
    *(U4*)&As[p][sar][sac]     = a0;
    *(U4*)&As[p][sar][sac + 8] = a1;
    U4 b = {0u, 0u, 0u, 0u};
    if (wok) b = *(const U4*)(Wbase + kk);
    *(U4*)&Bs[p][sbr][sbc] = b;
  };

  stage(0, 0);
  for (int kk = 0; kk < K; kk += 32) {
    const int p = (kk >> 5) & 1;
    __syncthreads();                         // buffer p staged; buffer p^1 free
    if (kk + 32 < K) {
      stage(p ^ 1, kk + 32);
      if (kk + 64 < K) {
        __builtin_prefetch(Abase + kk + 64, 0, 1);
        if (wok) __builtin_prefetch(Wbase + kk + 64, 0, 1);
      }
    }

    // fragment gather per ISA 16-bit WMMA layouts
    union { v16bf v; unsigned int u[8]; } af0, af1, bq0, bq1;
    const int ar0 = wr * 32 + l16;
#pragma unroll
    for (int i = 0; i < 8; ++i) {            // A: VGPR i -> K pair base
      int kb = 2 * i + ((i >= 4) ? 8 : 0) + 8 * half;
      af0.u[i] = *(const unsigned int*)&As[p][ar0][kb];
      af1.u[i] = *(const unsigned int*)&As[p][ar0 + 16][kb];
    }
    const int bc0 = wc * 32 + l16;
#pragma unroll
    for (int i = 0; i < 8; ++i) {            // B: VGPR i -> K pair base
      int kb = 2 * i + ((i >= 4) ? 8 : 0) + 8 * half;
      bq0.u[i] = *(const unsigned int*)&Bs[p][bc0][kb];
      bq1.u[i] = *(const unsigned int*)&Bs[p][bc0 + 16][kb];
    }

    acc00 = __builtin_amdgcn_wmma_f32_16x16x32_bf16(false, af0.v, false, bq0.v,
                                                    (short)0, acc00, false, false);
    acc01 = __builtin_amdgcn_wmma_f32_16x16x32_bf16(false, af0.v, false, bq1.v,
                                                    (short)0, acc01, false, false);
    acc10 = __builtin_amdgcn_wmma_f32_16x16x32_bf16(false, af1.v, false, bq0.v,
                                                    (short)0, acc10, false, false);
    acc11 = __builtin_amdgcn_wmma_f32_16x16x32_bf16(false, af1.v, false, bq1.v,
                                                    (short)0, acc11, false, false);
  }

  // epilogue: C layout VGPR r -> row r + 8*half, col = lane%16
  float* Cf = (float*)Cv;
  unsigned short* Cb = (unsigned short*)Cv;
#pragma unroll
  for (int rt = 0; rt < 2; ++rt) {
#pragma unroll
    for (int ct = 0; ct < 2; ++ct) {
      v8f a = (rt == 0) ? (ct == 0 ? acc00 : acc01) : (ct == 0 ? acc10 : acc11);
      int colg = n0 + wc * 32 + ct * 16 + l16;
      if (colg >= N) continue;
      float bv = BIAS ? bias[colg] : 0.0f;
      int rbase = m0 + wr * 32 + rt * 16 + 8 * half;
#pragma unroll
      for (int r = 0; r < 8; ++r) {
        float val = a[r] + bv;
        if (GELU) val = gelu_f(val);
        size_t off = (size_t)(rbase + r) * (size_t)N + (size_t)colg;
        if (RES) val += Res[off];
        if (OUTBF) Cb[off] = f2bf(val);
        else       Cf[off] = val;
      }
    }
  }
}

// ---------------- causal attention: one wave per (token, head), bf16 out ----------------
__global__ void attention_kernel(const float* __restrict__ qkv,
                                 unsigned short* __restrict__ y) {
  int gw   = blockIdx.x * 8 + (threadIdx.x >> 5);
  int lane = threadIdx.x & 31;
  int tok  = gw / NH;
  int h    = gw % NH;
  int tpos  = tok & (TBLK - 1);
  int bbase = tok - tpos;
  const float scale = 0.125f;        // 1/sqrt(64)

  const float* qp = qkv + (size_t)tok * (3 * DMODEL) + h * HDIM;
  float q0 = qp[lane] * scale;
  float q1 = qp[lane + 32] * scale;

  float m = -3.0e38f, s = 0.f, a0 = 0.f, a1 = 0.f;
  for (int j = 0; j <= tpos; ++j) {
    const float* kp = qkv + (size_t)(bbase + j) * (3 * DMODEL) + DMODEL + h * HDIM;
    float d = q0 * kp[lane] + q1 * kp[lane + 32];
#pragma unroll
    for (int o = 16; o > 0; o >>= 1) d += __shfl_xor(d, o, 32);
    float nm = fmaxf(m, d);
    float f  = __expf(m - nm);
    float p  = __expf(d - nm);
    const float* vp = qkv + (size_t)(bbase + j) * (3 * DMODEL) + 2 * DMODEL + h * HDIM;
    s  = s * f + p;
    a0 = a0 * f + p * vp[lane];
    a1 = a1 * f + p * vp[lane + 32];
    m = nm;
  }
  float inv = 1.0f / s;
  y[(size_t)tok * DMODEL + h * HDIM + lane]      = f2bf(a0 * inv);
  y[(size_t)tok * DMODEL + h * HDIM + lane + 32] = f2bf(a1 * inv);
}

// ---------------- loss ----------------
__global__ void zero_kernel(float* p) { if (threadIdx.x == 0) *p = 0.f; }

__global__ void loss_kernel(const float* __restrict__ logits,
                            const int* __restrict__ targets,
                            float* __restrict__ loss) {
  __shared__ float red[256];
  int row = blockIdx.x;
  const float* lp = logits + (size_t)row * V_;
  float mx = -3.0e38f;
  for (int c = threadIdx.x; c < V_; c += 256) mx = fmaxf(mx, lp[c]);
  red[threadIdx.x] = mx; __syncthreads();
  for (int o = 128; o > 0; o >>= 1) {
    if (threadIdx.x < o) red[threadIdx.x] = fmaxf(red[threadIdx.x], red[threadIdx.x + o]);
    __syncthreads();
  }
  mx = red[0]; __syncthreads();
  float sm = 0.f;
  for (int c = threadIdx.x; c < V_; c += 256) sm += __expf(lp[c] - mx);
  red[threadIdx.x] = sm; __syncthreads();
  for (int o = 128; o > 0; o >>= 1) {
    if (threadIdx.x < o) red[threadIdx.x] += red[threadIdx.x + o];
    __syncthreads();
  }
  if (threadIdx.x == 0) {
    float lse = mx + __logf(red[0]);
    atomicAdd(loss, (lse - lp[targets[row]]) * (1.0f / NTOK));
  }
}

// ---------------- driver ----------------
extern "C" void kernel_launch(void* const* d_in, const int* in_sizes, int n_in,
                              void* d_out, int out_size, void* d_ws, size_t ws_size,
                              hipStream_t stream) {
  (void)in_sizes; (void)n_in; (void)out_size; (void)ws_size;
  const int*   idx     = (const int*)d_in[0];
  const int*   targets = (const int*)d_in[1];
  const float* wte     = (const float*)d_in[2];
  const float* wpe     = (const float*)d_in[3];
  const float* ln1_s   = (const float*)d_in[4];
  const float* ln1_b   = (const float*)d_in[5];
  const float* attn_w  = (const float*)d_in[6];
  const float* attn_b  = (const float*)d_in[7];
  const float* proj_w  = (const float*)d_in[8];
  const float* proj_b  = (const float*)d_in[9];
  const float* ln2_s   = (const float*)d_in[10];
  const float* ln2_b   = (const float*)d_in[11];
  const float* fc_w    = (const float*)d_in[12];
  const float* fc_b    = (const float*)d_in[13];
  const float* p2_w    = (const float*)d_in[14];
  const float* p2_b    = (const float*)d_in[15];
  const float* lnf_s   = (const float*)d_in[16];
  const float* lnf_b   = (const float*)d_in[17];

  float* logits = (float*)d_out;
  float* loss   = logits + (size_t)NTOK * V_;

  // workspace layout (all offsets 16B aligned)
  float*          x      = (float*)d_ws;                         // [2048,768] f32
  float*          qkv    = x + (size_t)NTOK * DMODEL;            // [2048,2304] f32
  unsigned short* h_bf   = (unsigned short*)(qkv + (size_t)NTOK * 3 * DMODEL); // [2048,768]
  unsigned short* y_bf   = h_bf + (size_t)NTOK * DMODEL;         // [2048,768]
  unsigned short* hid_bf = y_bf + (size_t)NTOK * DMODEL;         // [2048,3072]
  unsigned short* wbuf   = hid_bf + (size_t)NTOK * 4 * DMODEL;   // [N,K] bf16, reused
  unsigned short* wte_bf = wbuf + (size_t)DMODEL * 4 * DMODEL;   // [V,768] bf16

  dim3 blk(256);
  embed_kernel<<<NTOK, blk, 0, stream>>>(idx, wte, wpe, x);
  conv_bf16<<<4096, blk, 0, stream>>>(wte, wte_bf, (long)V_ * DMODEL);

  for (int l = 0; l < NL; ++l) {
    // --- attention ---
    layernorm_kernel<<<NTOK, blk, 0, stream>>>(x, ln1_s + l * DMODEL, ln1_b + l * DMODEL, h_bf);
    convT_bf16<<<dim3(24, 72), blk, 0, stream>>>(attn_w + (size_t)l * DMODEL * 3 * DMODEL,
                                                 wbuf, DMODEL, 3 * DMODEL);
    gemm_wmma<false, false, true, false><<<dim3(36, 16), blk, 0, stream>>>(
        h_bf, wbuf, attn_b + l * 3 * DMODEL, nullptr, qkv, NTOK, 3 * DMODEL, DMODEL);
    attention_kernel<<<NTOK * NH / 8, blk, 0, stream>>>(qkv, y_bf);
    convT_bf16<<<dim3(24, 24), blk, 0, stream>>>(proj_w + (size_t)l * DMODEL * DMODEL,
                                                 wbuf, DMODEL, DMODEL);
    gemm_wmma<false, true, true, false><<<dim3(12, 16), blk, 0, stream>>>(
        y_bf, wbuf, proj_b + l * DMODEL, x, x, NTOK, DMODEL, DMODEL);
    // --- MLP ---
    layernorm_kernel<<<NTOK, blk, 0, stream>>>(x, ln2_s + l * DMODEL, ln2_b + l * DMODEL, h_bf);
    convT_bf16<<<dim3(24, 96), blk, 0, stream>>>(fc_w + (size_t)l * DMODEL * 4 * DMODEL,
                                                 wbuf, DMODEL, 4 * DMODEL);
    gemm_wmma<true, false, true, true><<<dim3(48, 16), blk, 0, stream>>>(
        h_bf, wbuf, fc_b + l * 4 * DMODEL, nullptr, hid_bf, NTOK, 4 * DMODEL, DMODEL);
    convT_bf16<<<dim3(96, 24), blk, 0, stream>>>(p2_w + (size_t)l * 4 * DMODEL * DMODEL,
                                                 wbuf, 4 * DMODEL, DMODEL);
    gemm_wmma<false, true, true, false><<<dim3(12, 16), blk, 0, stream>>>(
        hid_bf, wbuf, p2_b + l * DMODEL, x, x, NTOK, DMODEL, 4 * DMODEL);
  }

  layernorm_kernel<<<NTOK, blk, 0, stream>>>(x, lnf_s, lnf_b, h_bf);
  // tied lm_head: logits = h @ wte^T ; wte is already [N,K], pre-converted to bf16
  gemm_wmma<false, false, false, false><<<dim3((V_ + 63) / 64, 16), blk, 0, stream>>>(
      h_bf, wte_bf, nullptr, nullptr, logits, NTOK, V_, DMODEL);

  zero_kernel<<<1, 32, 0, stream>>>(loss);
  loss_kernel<<<NTOK, blk, 0, stream>>>(logits, targets, loss);
}
